// BertRNNWithSpeakerAttention_39651138077113
// MI455X (gfx1250) — compile-verified, matrix operand order checked by reading, not run on hardware
//
#include <hip/hip_runtime.h>
#include <cstdint>

// ---------------------------------------------------------------------------
// Types for CDNA5 WMMA (wave32): D(16x16 f32) = A(16x32 bf16) x B(32x16 bf16) + C
// ---------------------------------------------------------------------------
typedef __bf16 v16bf __attribute__((ext_vector_type(16)));
typedef float  v8f   __attribute__((ext_vector_type(8)));

__device__ __forceinline__ v8f zero8() {
  v8f z = {0.f,0.f,0.f,0.f,0.f,0.f,0.f,0.f};
  return z;
}

__device__ __forceinline__ v8f wmma_bf16(v16bf a, v16bf b, v8f c) {
  // (neg_a, A, neg_b, B, c_mod, C, reuse_a, reuse_b)
  return __builtin_amdgcn_wmma_f32_16x16x32_bf16(false, a, false, b, (short)0, c, false, false);
}

// Load a 16x32 bf16 fragment (A-layout / B-as-rows-of-W layout) from a tile
// whose K-pairs are contiguous in memory.  ISA 7.12.2:
//   lane l: row = l&15, half = l>>4
//   reg p (p<4): K = half*8 + 2p,2p+1 ; reg p (p>=4): K = 16 + half*8 + 2(p-4)..
__device__ __forceinline__ v16bf frag_ld(const __bf16* tile, int ld, int row0, int kbase) {
  int lane = threadIdx.x & 31;
  int r  = lane & 15;
  int hf = lane >> 4;
  const __bf16* rp = tile + (size_t)(row0 + r) * (size_t)ld + kbase + hf * 8;
  union { v16bf v; uint32_t u[8]; } f;
#pragma unroll
  for (int p = 0; p < 8; ++p) {
    int k = ((p & 4) << 2) + ((p & 3) << 1);   // 0,2,4,6,16,18,20,22
    f.u[p] = *(const uint32_t*)(rp + k);
  }
  return f.v;
}

// LDS byte offset of a generic pointer to __shared__ (ISA 10.2: LDS aperture
// address low 32 bits == LDS offset).
__device__ __forceinline__ unsigned lds_off(const void* p) {
  return (unsigned)(uintptr_t)p;
}

// Async global -> LDS copy of 16 bytes per lane (tracked by ASYNCcnt).
__device__ __forceinline__ void async_ld_b128(unsigned lds_addr, const void* gaddr) {
  asm volatile("global_load_async_to_lds_b128 %0, %1, off"
               :: "v"(lds_addr), "v"(gaddr) : "memory");
}

// ---------------------------------------------------------------------------
// Problem constants
// ---------------------------------------------------------------------------
#define NB    32
#define NT    512
#define NHID  768
#define NHEADS 4
#define HDIM  192         // 12 n-tiles of 16, 6 k-frags of 32
#define HID   384
#define G3    1152        // 3*HID
#define NCLS  43
#define NROWS (NB*NT)     // 16384

// ---------------------------------------------------------------------------
// f32 -> bf16 convert
// ---------------------------------------------------------------------------
__global__ void k_cvt(const float* __restrict__ src, __bf16* __restrict__ dst, long long n) {
  long long i = (long long)blockIdx.x * blockDim.x + threadIdx.x;
  if (i < n) dst[i] = (__bf16)src[i];
}

// ---------------------------------------------------------------------------
// Reverse rows per batch: xr[b][t] = x[b][clip(lens[b]-1-t,0,T-1)]   (8 bf16 / thread)
// ---------------------------------------------------------------------------
__global__ void k_rev(const __bf16* __restrict__ x, __bf16* __restrict__ xr,
                      const int* __restrict__ lens) {
  long long i = (long long)blockIdx.x * blockDim.x + threadIdx.x;
  const long long total = (long long)NROWS * (NHID / 8);
  if (i >= total) return;
  int g = (int)(i % (NHID / 8));
  long long bt = i / (NHID / 8);
  int t = (int)(bt % NT);
  int b = (int)(bt / NT);
  int s = lens[b] - 1 - t;
  s = s < 0 ? 0 : (s > NT - 1 ? NT - 1 : s);
  const uint4* sp = (const uint4*)(x + ((size_t)(b * NT + s)) * NHID) + g;
  uint4* dp = (uint4*)(xr + ((size_t)(b * NT + t)) * NHID) + g;
  *dp = *sp;
}

// ---------------------------------------------------------------------------
// Residual epilogue: xb = bf16( emb + (has_hist ? ctx_proj : 0) )
// ---------------------------------------------------------------------------
__global__ void k_resid(const float* __restrict__ emb, const float* __restrict__ cproj,
                        const unsigned* __restrict__ hist, __bf16* __restrict__ xb) {
  long long i = (long long)blockIdx.x * blockDim.x + threadIdx.x;
  const long long n = (long long)NROWS * NHID;
  if (i >= n) return;
  long long row = i / NHID;
  float v = emb[i] + (hist[row] ? cproj[i] : 0.f);
  xb[i] = (__bf16)v;
}

// ---------------------------------------------------------------------------
// Tiled bf16 GEMM:  C[M,N] = A[M,K] @ W[N,K]^T + bias
//  - 256 threads = 8 waves, block tile 128x128, K-step 32
//  - double-buffered GLOBAL_LOAD_ASYNC_TO_LDS_B128 staging (ASYNCcnt pipeline):
//    4 async issues/thread/stage; in-order completion => s_wait_asynccnt 4
//    guarantees the older stage landed while the newer one is in flight.
//  - wave (wm 0..3, wn 0..1) computes 32x64 = 2x4 WMMA tiles
//  - optional f32 and/or bf16 outputs; N may be ragged (fc: N=43)
// ---------------------------------------------------------------------------
__global__ __launch_bounds__(256)
void k_gemm(const __bf16* __restrict__ A, const __bf16* __restrict__ W,
            const float* __restrict__ bias, float* Cf, __bf16* Cb,
            int M, int N, int K) {
  __shared__ __bf16 As[2][128][40];
  __shared__ __bf16 Bs[2][128][40];
  int m0 = blockIdx.x * 128, n0 = blockIdx.y * 128;
  int tid = threadIdx.x, lane = tid & 31, w = tid >> 5;
  int wm = w >> 1, wn = w & 1;
  int r = lane & 15, hf = lane >> 4;

  v8f acc[2][4];
#pragma unroll
  for (int mt = 0; mt < 2; ++mt)
#pragma unroll
    for (int nt = 0; nt < 4; ++nt) acc[mt][nt] = zero8();

  int lr = tid >> 1;
  int lc = (tid & 1) * 16;

  // stage K-tile k0 into LDS buffer `buf` (async, no VGPR round-trip)
  auto stage = [&](int buf, int k0) {
    const __bf16* ga = A + (size_t)(m0 + lr) * K + k0 + lc;
    unsigned la = lds_off(&As[buf][lr][lc]);
    async_ld_b128(la,       ga);
    async_ld_b128(la + 16u, ga + 8);
    __builtin_prefetch(ga + 32, 0, 0);            // global_prefetch_b8, tile after next
    int nrow = n0 + lr;
    if (nrow < N) {
      const __bf16* gb = W + (size_t)nrow * K + k0 + lc;
      unsigned lb = lds_off(&Bs[buf][lr][lc]);
      async_ld_b128(lb,       gb);
      async_ld_b128(lb + 16u, gb + 8);
      __builtin_prefetch(gb + 32, 0, 0);
    } else {
      uint4 z = {0u, 0u, 0u, 0u};
      *(uint4*)&Bs[buf][lr][lc]     = z;          // tracked ds_store; covered by barrier
      *(uint4*)&Bs[buf][lr][lc + 8] = z;
    }
  };

  stage(0, 0);
  for (int k0 = 0; k0 < K; k0 += 32) {
    int cur = (k0 >> 5) & 1;
    bool more = (k0 + 32 < K);
    if (more) stage(cur ^ 1, k0 + 32);            // issue next stage before waiting
    if (more) asm volatile("s_wait_asynccnt 0x4" ::: "memory");
    else      asm volatile("s_wait_asynccnt 0x0" ::: "memory");
    __syncthreads();

    v16bf af[2], bf[4];
#pragma unroll
    for (int mt = 0; mt < 2; ++mt)
      af[mt] = frag_ld(&As[cur][0][0], 40, wm * 32 + mt * 16, 0);
#pragma unroll
    for (int nt = 0; nt < 4; ++nt)
      bf[nt] = frag_ld(&Bs[cur][0][0], 40, wn * 64 + nt * 16, 0);
#pragma unroll
    for (int mt = 0; mt < 2; ++mt)
#pragma unroll
      for (int nt = 0; nt < 4; ++nt) acc[mt][nt] = wmma_bf16(af[mt], bf[nt], acc[mt][nt]);
    __syncthreads();
  }

  // Epilogue: D layout -> lane holds rows m = hf*8+v, col n = r
#pragma unroll
  for (int mt = 0; mt < 2; ++mt) {
#pragma unroll
    for (int nt = 0; nt < 4; ++nt) {
      int gn = n0 + wn * 64 + nt * 16 + r;
      if (gn >= N) continue;
      float bv = bias ? bias[gn] : 0.f;
      int gmb = m0 + wm * 32 + mt * 16 + hf * 8;
#pragma unroll
      for (int v = 0; v < 8; ++v) {
        float o = acc[mt][nt][v] + bv;
        size_t idx = (size_t)(gmb + v) * N + gn;
        if (Cf) Cf[idx] = o;
        if (Cb) Cb[idx] = (__bf16)o;
      }
    }
  }
}

// ---------------------------------------------------------------------------
// Flash attention with speaker + strict-causal mask.
//  grid: B*NHEADS*(T/64) blocks, 128 threads = 4 waves, each wave = one 16-row q tile.
//  qkv rows: [q(768) | k(768) | v(768)], head slice h*192.
// ---------------------------------------------------------------------------
__global__ __launch_bounds__(128)
void k_attn(const __bf16* __restrict__ qkv, const int* __restrict__ spk,
            __bf16* __restrict__ ctx, unsigned* __restrict__ hist) {
  __shared__ __bf16 Vt[HDIM][34];      // V chunk transposed: [d][key]
  __shared__ __bf16 Pl[4][16][34];     // per-wave softmax probs (A-layout tile)

  int bi = blockIdx.x;
  int qt = bi & 7;
  int h  = (bi >> 3) & 3;
  int b  = bi >> 5;
  int w = threadIdx.x >> 5, lane = threadIdx.x & 31;
  int r = lane & 15, hf = lane >> 4;
  int qbase = qt * 64 + w * 16;
  const float scale = 0.07216878364870322f;    // 1/sqrt(192)

  // Q fragments, kept in registers for the whole pass
  const __bf16* qtile = qkv + ((size_t)(b * NT + qbase)) * (3 * NHID) + h * HDIM;
  v16bf qf[6];
#pragma unroll
  for (int kk = 0; kk < 6; ++kk) qf[kk] = frag_ld(qtile, 3 * NHID, 0, kk * 32);

  int qspk[8];
#pragma unroll
  for (int v = 0; v < 8; ++v) qspk[v] = spk[b * NT + qbase + hf * 8 + v];

  float mi[8], li[8];
#pragma unroll
  for (int v = 0; v < 8; ++v) { mi[v] = -3e38f; li[v] = 0.f; }
  v8f acc[12];
#pragma unroll
  for (int nt = 0; nt < 12; ++nt) acc[nt] = zero8();

  int jcmax = (qt * 64 + 62) >> 5;   // last key chunk any wave in block needs
  for (int jc = 0; jc <= jcmax; ++jc) {
    int kb = jc << 5;
    // cooperative stage of V chunk, transposed (pairs along key become contiguous)
    for (int i = threadIdx.x; i < 32 * HDIM; i += 128) {
      int key = i / HDIM, d = i - key * HDIM;
      Vt[d][key] = qkv[((size_t)(b * NT + kb + key)) * (3 * NHID) + 1536 + h * HDIM + d];
    }
    __syncthreads();

    bool active = (kb <= qbase + 14);
    if (active) {
      // S = Q K^T for two 16-key tiles
      v8f s0 = zero8(), s1 = zero8();
      const __bf16* ktile = qkv + ((size_t)(b * NT + kb)) * (3 * NHID) + NHID + h * HDIM;
#pragma unroll
      for (int kk = 0; kk < 6; ++kk) {
        v16bf k0 = frag_ld(ktile, 3 * NHID, 0,  kk * 32);
        s0 = wmma_bf16(qf[kk], k0, s0);
      }
#pragma unroll
      for (int kk = 0; kk < 6; ++kk) {
        v16bf k1 = frag_ld(ktile, 3 * NHID, 16, kk * 32);
        s1 = wmma_bf16(qf[kk], k1, s1);
      }
      int ks0 = spk[b * NT + kb + r];
      int ks1 = spk[b * NT + kb + 16 + r];

      float fac[8], p0a[8], p1a[8];
#pragma unroll
      for (int v = 0; v < 8; ++v) {
        int q = qbase + hf * 8 + v;
        bool m0 = (ks0 == qspk[v]) && (kb + r      < q);
        bool m1 = (ks1 == qspk[v]) && (kb + 16 + r < q);
        float sv0 = s0[v] * scale, sv1 = s1[v] * scale;
        float mx = fmaxf(m0 ? sv0 : -3e38f, m1 ? sv1 : -3e38f);
#pragma unroll
        for (int d = 1; d < 16; d <<= 1) mx = fmaxf(mx, __shfl_xor(mx, d, 32));
        float mnew = fmaxf(mi[v], mx);
        float f = __expf(mi[v] - mnew);            // mi==mnew==-3e38 -> 1, li stays 0
        float p0 = m0 ? __expf(sv0 - mnew) : 0.f;
        float p1 = m1 ? __expf(sv1 - mnew) : 0.f;
        float ps = p0 + p1;
#pragma unroll
        for (int d = 1; d < 16; d <<= 1) ps += __shfl_xor(ps, d, 32);
        li[v] = li[v] * f + ps;
        mi[v] = mnew;
        fac[v] = f; p0a[v] = p0; p1a[v] = p1;
      }
#pragma unroll
      for (int nt = 0; nt < 12; ++nt)
#pragma unroll
        for (int v = 0; v < 8; ++v) acc[nt][v] *= fac[v];
      // store P tile (D layout -> row-major 16x32) to per-wave LDS
#pragma unroll
      for (int v = 0; v < 8; ++v) {
        Pl[w][hf * 8 + v][r]      = (__bf16)p0a[v];
        Pl[w][hf * 8 + v][16 + r] = (__bf16)p1a[v];
      }
    }
    __syncthreads();
    if (active) {
      v16bf pf = frag_ld(&Pl[w][0][0], 34, 0, 0);
#pragma unroll
      for (int nt = 0; nt < 12; ++nt) {
        v16bf vf = frag_ld(&Vt[0][0], 34, nt * 16, 0);
        acc[nt] = wmma_bf16(pf, vf, acc[nt]);
      }
    }
    __syncthreads();
  }

  // Normalize and write ctx (bf16); rows with no history -> 0
  float inv[8];
#pragma unroll
  for (int v = 0; v < 8; ++v) inv[v] = (li[v] > 0.f) ? (1.f / li[v]) : 0.f;
#pragma unroll
  for (int nt = 0; nt < 12; ++nt)
#pragma unroll
    for (int v = 0; v < 8; ++v)
      ctx[((size_t)(b * NT + qbase + hf * 8 + v)) * NHID + h * HDIM + nt * 16 + r] =
          (__bf16)(acc[nt][v] * inv[v]);
  if (h == 0 && r == 0)
#pragma unroll
    for (int v = 0; v < 8; ++v)
      hist[b * NT + qbase + hf * 8 + v] = (li[v] > 0.f) ? 1u : 0u;
}

// ---------------------------------------------------------------------------
// Persistent bidirectional GRU recurrence.
//  grid = 4 blocks: blockIdx = dir*2 + batch-half.  1024 threads = 32 waves.
//  Per step: G[16,1152] = h[16,384] @ w_hh^T via WMMA (h bf16 in LDS,
//  w_hh bf16 from L2), then f32 gate math, h kept in f32 + bf16 copies in LDS.
//  LDS: h32 24576B | hb 12544B | G 73728B | lens 64B  = 110912B (dynamic).
// ---------------------------------------------------------------------------
__global__ __launch_bounds__(1024)
void k_gru(const float* __restrict__ xp_f, const float* __restrict__ xp_b,
           const __bf16* __restrict__ whh, const float* __restrict__ bhh,
           const int* __restrict__ lens, float* __restrict__ xnext) {
  extern __shared__ char smem[];
  float*  h32    = (float*)smem;                          // [16][384]
  __bf16* hb     = (__bf16*)(smem + 16 * 384 * 4);        // [16][392]
  float*  G      = (float*)(smem + 16 * 384 * 4 + 16 * 392 * 2);   // [16][1152]
  int*    lens_s = (int*)(smem + 16 * 384 * 4 + 16 * 392 * 2 + 16 * 1152 * 4);

  int dir = blockIdx.x >> 1, bh = blockIdx.x & 1;
  int tid = threadIdx.x, lane = tid & 31, w = tid >> 5;
  int r = lane & 15, hf = lane >> 4;

  const float*  xp = dir ? xp_b : xp_f;
  const __bf16* W  = whh + (size_t)dir * G3 * HID;
  const float*  bi = bhh + dir * G3;

  for (int i = tid; i < 16 * 384; i += 1024) h32[i] = 0.f;
  for (int i = tid; i < 16 * 392; i += 1024) hb[i] = (__bf16)0.f;
  if (tid < 16) lens_s[tid] = lens[bh * 16 + tid];
  __syncthreads();

  for (int t = 0; t < NT; ++t) {
    // ---- GEMM phase: G = h @ w_hh^T  (72 N-tiles over 32 waves) ----
    for (int nt = w; nt < 72; nt += 32) {
      v8f acc = zero8();
      const __bf16* wt = W + (size_t)(nt * 16) * HID;
#pragma unroll
      for (int kk = 0; kk < 12; ++kk) {
        v16bf a = frag_ld(hb, 392, 0, kk * 32);
        v16bf bfr = frag_ld(wt, HID, 0, kk * 32);
        acc = wmma_bf16(a, bfr, acc);
      }
      int n0 = nt * 16;
#pragma unroll
      for (int v = 0; v < 8; ++v) G[(hf * 8 + v) * G3 + n0 + r] = acc[v];
    }
    __syncthreads();
    // ---- gate phase ----
    for (int i = tid; i < 16 * 384; i += 1024) {
      int bl = i / 384;
      int u  = i - bl * 384;
      int b  = bh * 16 + bl;
      const float* xpp = xp + ((size_t)(b * NT + t)) * G3 + u;
      float xrv = xpp[0], xzv = xpp[384], xnv = xpp[768];
      float Gr = G[bl * G3 + u]        + bi[u];
      float Gz = G[bl * G3 + 384 + u]  + bi[384 + u];
      float Gn = G[bl * G3 + 768 + u]  + bi[768 + u];
      float rr = 1.f / (1.f + __expf(-(xrv + Gr)));
      float zz = 1.f / (1.f + __expf(-(xzv + Gz)));
      float nn = tanhf(xnv + rr * Gn);
      float hold = h32[i];
      float hnew = (1.f - zz) * nn + zz * hold;
      bool vld = (t < lens_s[bl]);
      float hv = vld ? hnew : hold;
      h32[i] = hv;
      hb[bl * 392 + u] = (__bf16)hv;
      if (dir == 0) {
        xnext[((size_t)(b * NT + t)) * NHID + u] = vld ? hnew : 0.f;
      } else if (vld) {
        int to = lens_s[bl] - 1 - t;
        xnext[((size_t)(b * NT + to)) * NHID + HID + u] = hnew;
      }
    }
    __syncthreads();
  }
}

// ---------------------------------------------------------------------------
// Host-side launcher
// ---------------------------------------------------------------------------
static inline size_t alignup(size_t x) { return (x + 255) & ~(size_t)255; }

extern "C" void kernel_launch(void* const* d_in, const int* in_sizes, int n_in,
                              void* d_out, int out_size, void* d_ws, size_t ws_size,
                              hipStream_t stream) {
  (void)in_sizes; (void)n_in; (void)out_size; (void)ws_size;

  const float* emb   = (const float*)d_in[0];
  const int*   spk   = (const int*)d_in[1];
  const int*   lens  = (const int*)d_in[2];
  const float* w_qkv = (const float*)d_in[3];
  const float* b_qkv = (const float*)d_in[4];
  const float* w_out = (const float*)d_in[5];
  const float* b_out = (const float*)d_in[6];
  const float* w_ih  = (const float*)d_in[7];
  const float* w_hh  = (const float*)d_in[8];
  const float* b_ih  = (const float*)d_in[9];
  const float* b_hh  = (const float*)d_in[10];
  const float* fc_w  = (const float*)d_in[11];
  const float* fc_b  = (const float*)d_in[12];
  float* out = (float*)d_out;

  // workspace carve-up
  char* ws = (char*)d_ws;
  size_t off = 0;
  auto take = [&](size_t bytes) { char* p = ws + off; off = alignup(off + bytes); return p; };
  __bf16* xb    = (__bf16*)take((size_t)NROWS * NHID * 2);
  __bf16* qkvb  = (__bf16*)take((size_t)NROWS * 3 * NHID * 2);
  __bf16* ctxb  = (__bf16*)take((size_t)NROWS * NHID * 2);
  float*  cproj = (float*) take((size_t)NROWS * NHID * 4);
  float*  xnext = (float*) take((size_t)NROWS * NHID * 4);
  float*  xpf   = (float*) take((size_t)NROWS * G3 * 4);
  float*  xpb   = (float*) take((size_t)NROWS * G3 * 4);
  __bf16* xrev  = (__bf16*)take((size_t)NROWS * NHID * 2);
  unsigned* hist= (unsigned*)take((size_t)NROWS * 4);
  __bf16* wqkvb = (__bf16*)take((size_t)3 * NHID * NHID * 2);
  __bf16* woutb = (__bf16*)take((size_t)NHID * NHID * 2);
  __bf16* wihb  = (__bf16*)take((size_t)4 * G3 * NHID * 2);
  __bf16* whhb  = (__bf16*)take((size_t)4 * G3 * HID * 2);
  __bf16* fcwb  = (__bf16*)take((size_t)NCLS * NHID * 2);

  auto cvt = [&](const float* s, __bf16* d, long long n) {
    k_cvt<<<(unsigned)((n + 255) / 256), 256, 0, stream>>>(s, d, n);
  };

  // weight / input conversions to bf16
  cvt(emb,   xb,    (long long)NROWS * NHID);
  cvt(w_qkv, wqkvb, (long long)3 * NHID * NHID);
  cvt(w_out, woutb, (long long)NHID * NHID);
  cvt(w_ih,  wihb,  (long long)4 * G3 * NHID);
  cvt(w_hh,  whhb,  (long long)4 * G3 * HID);
  cvt(fc_w,  fcwb,  (long long)NCLS * NHID);

  // QKV projection: [16384,2304] bf16
  k_gemm<<<dim3(NROWS / 128, (3 * NHID) / 128), 256, 0, stream>>>(
      xb, wqkvb, b_qkv, (float*)nullptr, qkvb, NROWS, 3 * NHID, NHID);

  // attention
  k_attn<<<NB * NHEADS * (NT / 64), 128, 0, stream>>>(qkvb, spk, ctxb, hist);

  // output projection (f32), then residual+mask -> xb (bf16 GRU input)
  k_gemm<<<dim3(NROWS / 128, NHID / 128), 256, 0, stream>>>(
      ctxb, woutb, b_out, cproj, (__bf16*)nullptr, NROWS, NHID, NHID);
  {
    long long n = (long long)NROWS * NHID;
    k_resid<<<(unsigned)((n + 255) / 256), 256, 0, stream>>>(emb, cproj, hist, xb);
  }

  // GRU layers
  const size_t gru_smem = 16 * 384 * 4 + 16 * 392 * 2 + 16 * 1152 * 4 + 64; // 110912
  (void)hipFuncSetAttribute((const void*)k_gru,
                            hipFuncAttributeMaxDynamicSharedMemorySize, (int)gru_smem);
  for (int l = 0; l < 2; ++l) {
    const __bf16* wih_f = wihb + (size_t)(l * 2 + 0) * G3 * NHID;
    const __bf16* wih_b = wihb + (size_t)(l * 2 + 1) * G3 * NHID;
    const float*  bih_f = b_ih + (size_t)(l * 2 + 0) * G3;
    const float*  bih_b = b_ih + (size_t)(l * 2 + 1) * G3;
    // xp_fwd = x @ w_ih^T + b_ih
    k_gemm<<<dim3(NROWS / 128, G3 / 128), 256, 0, stream>>>(
        xb, wih_f, bih_f, xpf, (__bf16*)nullptr, NROWS, G3, NHID);
    // reversed input, then xp_bwd
    {
      long long n = (long long)NROWS * (NHID / 8);
      k_rev<<<(unsigned)((n + 255) / 256), 256, 0, stream>>>(xb, xrev, lens);
    }
    k_gemm<<<dim3(NROWS / 128, G3 / 128), 256, 0, stream>>>(
        xrev, wih_b, bih_b, xpb, (__bf16*)nullptr, NROWS, G3, NHID);
    // recurrence (fwd+bwd, two batch halves) -> xnext (pre-zeroed for masking)
    hipMemsetAsync(xnext, 0, (size_t)NROWS * NHID * 4, stream);
    k_gru<<<4, 1024, gru_smem, stream>>>(
        xpf, xpb, whhb + (size_t)l * 2 * G3 * HID, b_hh + (size_t)l * 2 * G3,
        lens, xnext);
    // next-layer bf16 input
    cvt(xnext, xb, (long long)NROWS * NHID);
  }

  // final FC: [16384,43] f32 into d_out (row order = (b,t,c), matching reference)
  k_gemm<<<dim3(NROWS / 128, 1), 256, 0, stream>>>(
      xb, fcwb, fc_b, out, (__bf16*)nullptr, NROWS, NCLS, NHID);
}
